// LocalSelfAttentionEncoderLayer_23957327577715
// MI455X (gfx1250) — matrix operand order repaired and checked by reading, not compiled, not run on hardware
//
#include <hip/hip_runtime.h>
#include <hip/hip_bf16.h>

// ---------------------------------------------------------------------------
// Types for CDNA5 WMMA (wave32)
// ---------------------------------------------------------------------------
typedef __attribute__((ext_vector_type(16))) __bf16 v16bf;
typedef __attribute__((ext_vector_type(8)))  float  v8f;

union FragU {
    v16bf v;
    uint4 q[2];
    unsigned short s[16];
};

__device__ __forceinline__ unsigned short f2bf(float f) {
    union { float f; unsigned u; } c; c.f = f;
    unsigned r = c.u + 0x7FFFu + ((c.u >> 16) & 1u);   // round-to-nearest-even
    return (unsigned short)(r >> 16);
}

// A fragment: 16x32 bf16, rows of A (M x K, row-major, ld = lda elements).
// Lane layout used by this code: lane m = lane&15 holds row m; lanes 0-15 hold
// K = 0..15, lanes 16-31 hold K = 16..31 (contiguous per lane).
__device__ __forceinline__ v16bf load_a_frag(const unsigned short* A, int lda,
                                             int row0, int k0, int lane) {
    int m  = lane & 15;
    int hi = lane >> 4;
    const unsigned short* p = A + (size_t)(row0 + m) * lda + k0 + hi * 16;
    FragU f;
    f.q[0] = *(const uint4*)(p);
    f.q[1] = *(const uint4*)(p + 8);
    return f.v;
}

// B fragment: 32x16 bf16 (K x N).  Column n of B is a K-contiguous row of the
// (N x K, row-major) weight matrix.  Lane n = lane&15 holds column n; lanes
// 0-15 hold K=0..15, lanes 16-31 hold K=16..31 (contiguous per lane).
__device__ __forceinline__ v16bf load_b_frag(const unsigned short* Bw, int ldb,
                                             int col0, int k0, int lane) {
    int n  = lane & 15;
    int hi = lane >> 4;
    const unsigned short* p = Bw + (size_t)(col0 + n) * ldb + k0 + hi * 16;
    FragU f;
    f.q[0] = *(const uint4*)(p);
    f.q[1] = *(const uint4*)(p + 8);
    return f.v;
}

__device__ __forceinline__ v8f wmma_bf16(v16bf a, v16bf b, v8f c) {
    return __builtin_amdgcn_wmma_f32_16x16x32_bf16(false, a, false, b,
                                                   (short)0, c, false, false);
}

// ---------------------------------------------------------------------------
// fp32 -> bf16 conversion (vectorized: 8 elements / thread, 128-bit stores)
// ---------------------------------------------------------------------------
__global__ void cvt_f32_bf16(const float* __restrict__ in,
                             unsigned short* __restrict__ out, int n) {
    int idx = (blockIdx.x * 256 + threadIdx.x) * 8;
    if (idx + 8 <= n) {
        float4 a = *(const float4*)(in + idx);
        float4 b = *(const float4*)(in + idx + 4);
        uint4 o;
        o.x = (unsigned)f2bf(a.x) | ((unsigned)f2bf(a.y) << 16);
        o.y = (unsigned)f2bf(a.z) | ((unsigned)f2bf(a.w) << 16);
        o.z = (unsigned)f2bf(b.x) | ((unsigned)f2bf(b.y) << 16);
        o.w = (unsigned)f2bf(b.z) | ((unsigned)f2bf(b.w) << 16);
        *(uint4*)(out + idx) = o;
    } else {
        for (int i = idx; i < n; ++i) out[i] = f2bf(in[i]);
    }
}

// ---------------------------------------------------------------------------
// GEMM: C[M,N] = A[M,K] * Bw[N,K]^T + bias  (optional ReLU)
// A, Bw in bf16 (row-major, K contiguous), accum fp32.
// Output fp32 (Cf) and/or bf16 (Cb), either may be null.
// Block = 256 threads = 8 waves; wave computes 32x64 (2 A-frags reuse each
// B-frag -> 8 WMMAs per K-step); block covers 256x64 output.
// ---------------------------------------------------------------------------
template <bool RELU>
__global__ __launch_bounds__(256)
void gemm_bf16_wmma(const unsigned short* __restrict__ A,
                    const unsigned short* __restrict__ Bw,
                    const float* __restrict__ bias,
                    float* __restrict__ Cf,
                    unsigned short* __restrict__ Cb,
                    int M, int N, int K) {
    int lane = threadIdx.x & 31;
    int wave = threadIdx.x >> 5;
    int row0 = blockIdx.y * 256 + wave * 32;
    int col0 = blockIdx.x * 64;

    v8f acc[2][4] = {{v8f{}, v8f{}, v8f{}, v8f{}},
                     {v8f{}, v8f{}, v8f{}, v8f{}}};

    for (int k0 = 0; k0 < K; k0 += 32) {
        // pull the K-step after next toward the WGP while WMMAs drain
        if (k0 + 64 < K) {
            __builtin_prefetch(A + (size_t)(row0 + (lane & 15)) * K + k0 + 64, 0, 0);
            __builtin_prefetch(Bw + (size_t)(col0 + (lane & 15) + (lane >> 4) * 16) * K + k0 + 64, 0, 0);
        }
        v16bf a0 = load_a_frag(A, K, row0,      k0, lane);
        v16bf a1 = load_a_frag(A, K, row0 + 16, k0, lane);
#pragma unroll
        for (int j = 0; j < 4; ++j) {
            v16bf b = load_b_frag(Bw, K, col0 + j * 16, k0, lane);
            acc[0][j] = wmma_bf16(a0, b, acc[0][j]);
            acc[1][j] = wmma_bf16(a1, b, acc[1][j]);
        }
    }

    int n  = lane & 15;
    int hi = lane >> 4;
#pragma unroll
    for (int rb = 0; rb < 2; ++rb) {
#pragma unroll
        for (int j = 0; j < 4; ++j) {
            int col = col0 + j * 16 + n;
            float bv = bias ? bias[col] : 0.0f;
#pragma unroll
            for (int r = 0; r < 8; ++r) {
                float v = acc[rb][j][r] + bv;
                if (RELU) v = fmaxf(v, 0.0f);
                size_t idx = (size_t)(row0 + rb * 16 + r + hi * 8) * N + col;
                if (Cf) Cf[idx] = v;
                if (Cb) Cb[idx] = f2bf(v);
            }
        }
    }
}

// ---------------------------------------------------------------------------
// Banded flash attention.  qkv: (B*S, 3D) bf16.  Output attn_b: (B*S, D) bf16.
// One wave handles one 16-query tile for one (b,h); streams 32-key blocks
// across the band [q0-128, q0+159] with online softmax.
// ---------------------------------------------------------------------------
#define SEQ   2048
#define DMODEL 1024
#define NHEAD 16
#define HDIM  64
#define WIN   128
#define LDQKV 3072

__global__ __launch_bounds__(256)
void attn_banded(const unsigned short* __restrict__ qkv,
                 unsigned short* __restrict__ attn_b) {
    __shared__ __align__(16) unsigned short plds[8][16 * 32];

    int lane = threadIdx.x & 31;
    int wave = threadIdx.x >> 5;
    int tile = blockIdx.x * 8 + wave;     // 4096 tiles total
    int qt = tile & 127;
    int h  = (tile >> 7) & 15;
    int b  = tile >> 11;
    int q0 = qt * 16;

    const unsigned short* Qp = qkv + (size_t)b * SEQ * LDQKV + h * HDIM;
    const unsigned short* Kp = Qp + DMODEL;
    const unsigned short* Vp = Qp + 2 * DMODEL;

    int m  = lane & 15;
    int hi = lane >> 4;

    // Q fragments over hd = 64 (two K-chunks of 32)
    v16bf qa0 = load_a_frag(Qp, LDQKV, q0, 0,  lane);
    v16bf qa1 = load_a_frag(Qp, LDQKV, q0, 32, lane);

    float mrow[8], lrow[8];
#pragma unroll
    for (int r = 0; r < 8; ++r) { mrow[r] = -1e30f; lrow[r] = 0.0f; }
    v8f o[4] = {v8f{}, v8f{}, v8f{}, v8f{}};

    const float scale = 0.125f;   // 1/sqrt(64)
    int kstart = q0 - WIN;

    for (int ks = 0; ks < 9; ++ks) {
        int kb = kstart + ks * 32;                 // 32-key block (may be OOB)

        // ---- scores: two 16x16 tiles (keys kb..+15, kb+16..+31) ----
        v8f s0 = v8f{}, s1 = v8f{};
        {
            int kc0 = kb;      kc0 = kc0 < 0 ? 0 : (kc0 > SEQ - 16 ? SEQ - 16 : kc0);
            int kc1 = kb + 16; kc1 = kc1 < 0 ? 0 : (kc1 > SEQ - 16 ? SEQ - 16 : kc1);
            v16bf bk;
            bk = load_b_frag(Kp, LDQKV, kc0, 0,  lane); s0 = wmma_bf16(qa0, bk, s0);
            bk = load_b_frag(Kp, LDQKV, kc0, 32, lane); s0 = wmma_bf16(qa1, bk, s0);
            bk = load_b_frag(Kp, LDQKV, kc1, 0,  lane); s1 = wmma_bf16(qa0, bk, s1);
            bk = load_b_frag(Kp, LDQKV, kc1, 32, lane); s1 = wmma_bf16(qa1, bk, s1);
        }

        // ---- mask + online softmax (rows live in (r,hi); cols in lane&15) ----
        int kj0 = kb + (lane & 15);
        int kj1 = kb + 16 + (lane & 15);
        float alpha[8];
#pragma unroll
        for (int r = 0; r < 8; ++r) {
            int qi = q0 + r + hi * 8;
            float v0 = s0[r] * scale;
            float v1 = s1[r] * scale;
            bool ok0 = (kj0 >= 0) && (kj0 < SEQ) && (kj0 >= qi - WIN) && (kj0 <= qi + WIN);
            bool ok1 = (kj1 >= 0) && (kj1 < SEQ) && (kj1 >= qi - WIN) && (kj1 <= qi + WIN);
            v0 = ok0 ? v0 : -1e30f;
            v1 = ok1 ? v1 : -1e30f;

            float mx = fmaxf(v0, v1);
            mx = fmaxf(mx, __shfl_xor(mx, 1));
            mx = fmaxf(mx, __shfl_xor(mx, 2));
            mx = fmaxf(mx, __shfl_xor(mx, 4));
            mx = fmaxf(mx, __shfl_xor(mx, 8));

            float mn = fmaxf(mrow[r], mx);
            float al = __expf(mrow[r] - mn);
            float p0 = ok0 ? __expf(v0 - mn) : 0.0f;
            float p1 = ok1 ? __expf(v1 - mn) : 0.0f;
            float ps = p0 + p1;
            ps += __shfl_xor(ps, 1);
            ps += __shfl_xor(ps, 2);
            ps += __shfl_xor(ps, 4);
            ps += __shfl_xor(ps, 8);

            lrow[r] = al * lrow[r] + ps;
            mrow[r] = mn;
            alpha[r] = al;

            // stage P tile (16 q x 32 k) to LDS in row-major bf16
            plds[wave][(r + 8 * hi) * 32 + (lane & 15)]      = f2bf(p0);
            plds[wave][(r + 8 * hi) * 32 + 16 + (lane & 15)] = f2bf(p1);
        }
#pragma unroll
        for (int j = 0; j < 4; ++j)
#pragma unroll
            for (int r = 0; r < 8; ++r) o[j][r] *= alpha[r];

        asm volatile("s_wait_dscnt 0" ::: "memory");

        // reload P as 16x32 A-fragment
        FragU pf;
        const unsigned short* pp = &plds[wave][0];
        pf.q[0] = *(const uint4*)(pp + m * 32 + hi * 16);
        pf.q[1] = *(const uint4*)(pp + m * 32 + hi * 16 + 8);

        // ---- O += P * V :  B fragment column n = hd col, K dim = 32 keys ----
        int kcb = kb + hi * 16;
#pragma unroll
        for (int j = 0; j < 4; ++j) {
            FragU vf;
#pragma unroll
            for (int kk = 0; kk < 16; ++kk) {
                int krow = kcb + kk;
                krow = krow < 0 ? 0 : (krow > SEQ - 1 ? SEQ - 1 : krow);
                vf.s[kk] = Vp[(size_t)krow * LDQKV + j * 16 + (lane & 15)];
            }
            o[j] = wmma_bf16(pf.v, vf.v, o[j]);
        }
    }

    // ---- normalize + store bf16 ----
#pragma unroll
    for (int j = 0; j < 4; ++j)
#pragma unroll
        for (int r = 0; r < 8; ++r) {
            int qi = q0 + r + hi * 8;
            float v = o[j][r] / lrow[r];
            attn_b[((size_t)b * SEQ + qi) * DMODEL + h * HDIM + j * 16 + (lane & 15)] = f2bf(v);
        }
}

// ---------------------------------------------------------------------------
// out = LayerNorm(xin + resid) * g + b   (fp32; optional bf16 copy)
// One block (256 threads) per row of D=1024.
// ---------------------------------------------------------------------------
__global__ __launch_bounds__(256)
void add_layernorm(const float* __restrict__ xin, const float* __restrict__ resid,
                   const float* __restrict__ g, const float* __restrict__ bta,
                   float* __restrict__ outf, unsigned short* __restrict__ outb) {
    __shared__ float red[8];
    const int D = 1024;
    int row = blockIdx.x;
    const float* xr = xin   + (size_t)row * D;
    const float* rr = resid + (size_t)row * D;

    float v[4];
    float s = 0.0f;
#pragma unroll
    for (int i = 0; i < 4; ++i) {
        int c = threadIdx.x + i * 256;
        v[i] = xr[c] + rr[c];
        s += v[i];
    }
    for (int off = 16; off >= 1; off >>= 1) s += __shfl_xor(s, off);
    if ((threadIdx.x & 31) == 0) red[threadIdx.x >> 5] = s;
    __syncthreads();
    float tot = 0.0f;
#pragma unroll
    for (int w = 0; w < 8; ++w) tot += red[w];
    float mu = tot * (1.0f / D);
    __syncthreads();

    float s2 = 0.0f;
#pragma unroll
    for (int i = 0; i < 4; ++i) { float d = v[i] - mu; s2 += d * d; }
    for (int off = 16; off >= 1; off >>= 1) s2 += __shfl_xor(s2, off);
    if ((threadIdx.x & 31) == 0) red[threadIdx.x >> 5] = s2;
    __syncthreads();
    float tot2 = 0.0f;
#pragma unroll
    for (int w = 0; w < 8; ++w) tot2 += red[w];
    float inv = rsqrtf(tot2 * (1.0f / D) + 1e-5f);

#pragma unroll
    for (int i = 0; i < 4; ++i) {
        int c = threadIdx.x + i * 256;
        float y = (v[i] - mu) * inv * g[c] + bta[c];
        outf[(size_t)row * D + c] = y;
        if (outb) outb[(size_t)row * D + c] = f2bf(y);
    }
}

// ---------------------------------------------------------------------------
// Host orchestration
// ---------------------------------------------------------------------------
extern "C" void kernel_launch(void* const* d_in, const int* in_sizes, int n_in,
                              void* d_out, int out_size, void* d_ws, size_t ws_size,
                              hipStream_t stream) {
    (void)in_sizes; (void)n_in; (void)out_size; (void)ws_size;

    const float* x    = (const float*)d_in[0];
    const float* wqkv = (const float*)d_in[1];
    const float* bqkv = (const float*)d_in[2];
    const float* wout = (const float*)d_in[3];
    const float* bout = (const float*)d_in[4];
    const float* ln1g = (const float*)d_in[5];
    const float* ln1b = (const float*)d_in[6];
    const float* w1   = (const float*)d_in[7];
    const float* b1   = (const float*)d_in[8];
    const float* w2   = (const float*)d_in[9];
    const float* b2   = (const float*)d_in[10];
    const float* ln2g = (const float*)d_in[11];
    const float* ln2b = (const float*)d_in[12];

    const int M = 2 * 2048;    // B*S rows
    const int D = 1024, DFF = 4096, D3 = 3072;

    char* ws = (char*)d_ws;
    size_t off = 0;
    auto alloc = [&](size_t bytes) -> void* {
        void* p = ws + off;
        off += (bytes + 255) & ~(size_t)255;
        return p;
    };
    unsigned short* xb    = (unsigned short*)alloc((size_t)M * D * 2);
    unsigned short* wqkvb = (unsigned short*)alloc((size_t)D3 * D * 2);
    unsigned short* woutb = (unsigned short*)alloc((size_t)D * D * 2);
    unsigned short* w1b   = (unsigned short*)alloc((size_t)DFF * D * 2);
    unsigned short* w2b   = (unsigned short*)alloc((size_t)D * DFF * 2);
    unsigned short* qkvb  = (unsigned short*)alloc((size_t)M * D3 * 2);
    unsigned short* attnb = (unsigned short*)alloc((size_t)M * D * 2);
    float*          projf = (float*)alloc((size_t)M * D * 4);
    float*          x2f   = (float*)alloc((size_t)M * D * 4);
    unsigned short* x2b   = (unsigned short*)alloc((size_t)M * D * 2);
    unsigned short* ff1b  = (unsigned short*)alloc((size_t)M * DFF * 2);
    float*          ff2f  = (float*)alloc((size_t)M * D * 4);

    auto cvt = [&](const float* in, unsigned short* out, int n) {
        cvt_f32_bf16<<<(n + 2047) / 2048, 256, 0, stream>>>(in, out, n);
    };
    cvt(x,    xb,    M * D);
    cvt(wqkv, wqkvb, D3 * D);
    cvt(wout, woutb, D * D);
    cvt(w1,   w1b,   DFF * D);
    cvt(w2,   w2b,   D * DFF);

    // QKV projection -> bf16 only
    gemm_bf16_wmma<false><<<dim3(D3 / 64, M / 256), 256, 0, stream>>>(
        xb, wqkvb, bqkv, nullptr, qkvb, M, D3, D);

    // banded attention -> bf16
    attn_banded<<<512, 256, 0, stream>>>(qkvb, attnb);

    // out projection -> fp32 (residual input)
    gemm_bf16_wmma<false><<<dim3(D / 64, M / 256), 256, 0, stream>>>(
        attnb, woutb, bout, projf, nullptr, M, D, D);

    // x2 = LN(x + attn_out)  (fp32 + bf16)
    add_layernorm<<<M, 256, 0, stream>>>(x, projf, ln1g, ln1b, x2f, x2b);

    // FFN1 + ReLU -> bf16 only
    gemm_bf16_wmma<true><<<dim3(DFF / 64, M / 256), 256, 0, stream>>>(
        x2b, w1b, b1, nullptr, ff1b, M, DFF, D);

    // FFN2 -> fp32
    gemm_bf16_wmma<false><<<dim3(D / 64, M / 256), 256, 0, stream>>>(
        ff1b, w2b, b2, ff2f, nullptr, M, D, DFF);

    // out = LN(x2 + ff)
    add_layernorm<<<M, 256, 0, stream>>>(x2f, ff2f, ln2g, ln2b, (float*)d_out, nullptr);
}
